// TFBertSelfAttention_87522843561354
// MI455X (gfx1250) — compile-verified, matrix operand order checked by reading, not run on hardware
//
#include <hip/hip_runtime.h>
#include <hip/hip_bf16.h>

// ---------------------------------------------------------------------------
// TFBertSelfAttention forward for MI455X (gfx1250, wave32, WMMA + async LDS).
// B=2, S=2048, D=768, H=12, DH=64.
// ---------------------------------------------------------------------------

#define BB 2
#define SS 2048
#define DD 768
#define HH 12
#define DHH 64
#define BS (BB * SS)             // 4096 rows of X
#define COLT (DD / 16)           // 48 col tiles
#define ROWT (BS / 16)           // 256 row tiles
#define GEMM_WAVES (3 * (ROWT / 2) * (COLT / 2))   // 9216 (2x2 tiles per wave)
#define CHUNK 64                 // keys per LDS chunk in attention
#define LPAD 72                  // padded LDS row: 144B = 36 dwords -> 0 bank conflicts

typedef __bf16 bf16_t;
typedef __attribute__((ext_vector_type(8)))  __bf16 v8bf;
typedef __attribute__((ext_vector_type(16))) __bf16 v16bf;
typedef __attribute__((ext_vector_type(8)))  float  v8f;

static __device__ inline v8f wmma_bf16(v16bf a, v16bf b, v8f c) {
    return __builtin_amdgcn_wmma_f32_16x16x32_bf16(
        false, a, false, b, (short)0, c, false, false);
}

// WMMA 16-bit fragment load from a row-major row pointer:
// elements j=0..7 -> base[koff+j], j=8..15 -> base[16+koff+j-8]; koff=8*(lane>=16)
static __device__ inline v16bf load_frag(const bf16_t* __restrict__ base, int koff) {
    v8bf a = *reinterpret_cast<const v8bf*>(base + koff);
    v8bf b = *reinterpret_cast<const v8bf*>(base + 16 + koff);
    v16bf r;
#pragma unroll
    for (int j = 0; j < 8; ++j) { r[j] = a[j]; r[j + 8] = b[j]; }
    return r;
}

// gfx1250 async copy: 16B global -> LDS, tracked by ASYNCcnt.
static __device__ inline void async_copy_b128(unsigned lds_off, const bf16_t* g) {
    asm volatile("global_load_async_to_lds_b128 %0, %1, off"
                 :: "v"(lds_off), "v"(g) : "memory");
}
static __device__ inline void wait_async4() {
    asm volatile("s_wait_asynccnt 0x4" ::: "memory");
}
static __device__ inline void wait_async0() {
    asm volatile("s_wait_asynccnt 0x0" ::: "memory");
}

// ---------------------------------------------------------------------------
// Kernel 0a: X fp32 -> bf16
__global__ void cvt_x_kernel(const float* __restrict__ X, bf16_t* __restrict__ Xb, int n) {
    int i = blockIdx.x * 256 + threadIdx.x;
    if (i < n) Xb[i] = (bf16_t)X[i];
}

// Kernel 0b: W[k][o] fp32 -> Wt[w][o][k] bf16
__global__ void cvt_w_kernel(const float* __restrict__ Wq, const float* __restrict__ Wk,
                             const float* __restrict__ Wv, bf16_t* __restrict__ Wt) {
    int i = blockIdx.x * 256 + threadIdx.x;
    if (i >= 3 * DD * DD) return;
    int w = i / (DD * DD);
    int rem = i - w * (DD * DD);
    int o = rem / DD, k = rem - o * DD;
    const float* W = (w == 0) ? Wq : (w == 1) ? Wk : Wv;
    Wt[i] = (bf16_t)W[k * DD + o];
}

// ---------------------------------------------------------------------------
// Kernel 1: QKV projection GEMM, 2x2 register-blocked tiles per wave.
// Q pre-scaled by (1/sqrt(DH))*log2(e). Q,K -> [B,H,S,DH]; V -> [B,H,DH,S].
static __device__ inline void store_tile(v8f c, int mat, int rowb, int col,
                                         float bcol, int koff,
                                         bf16_t* __restrict__ Qh,
                                         bf16_t* __restrict__ Kh,
                                         bf16_t* __restrict__ Vt) {
    const float qscale = 0.125f * 1.4426950408889634f;  // 1/sqrt(64)*log2(e)
    const int h = col >> 6, dh = col & 63;
#pragma unroll
    for (int rr = 0; rr < 8; ++rr) {
        const int grow = rowb + rr + koff;
        const int bidx = grow >> 11;
        const int sidx = grow & 2047;
        const int head = bidx * HH + h;
        float v = c[rr] + bcol;
        if (mat == 0) {
            Qh[((head * SS + sidx) << 6) + dh] = (bf16_t)(v * qscale);
        } else if (mat == 1) {
            Kh[((head * SS + sidx) << 6) + dh] = (bf16_t)v;
        } else {
            Vt[(head * DHH + dh) * SS + sidx] = (bf16_t)v;
        }
    }
}

__global__ __launch_bounds__(256) void qkv_gemm_kernel(
        const bf16_t* __restrict__ Xb, const bf16_t* __restrict__ Wt,
        const float* __restrict__ bq, const float* __restrict__ bk,
        const float* __restrict__ bv,
        bf16_t* __restrict__ Qh, bf16_t* __restrict__ Kh, bf16_t* __restrict__ Vt) {
    const int wid = (blockIdx.x * blockDim.x + threadIdx.x) >> 5;
    if (wid >= GEMM_WAVES) return;
    const int lane = threadIdx.x & 31;
    const int lo = lane & 15, hi = lane >> 4, koff = hi << 3;

    const int per_mat = (ROWT / 2) * (COLT / 2);      // 3072
    const int mat = wid / per_mat;
    const int t   = wid - mat * per_mat;
    const int tr  = t / (COLT / 2);
    const int tc  = t - tr * (COLT / 2);
    const int rowb = tr * 32, colb = tc * 32;

    const bf16_t* __restrict__ Wm = Wt + mat * (DD * DD);
    const bf16_t* __restrict__ a0 = Xb + (rowb + lo) * DD;
    const bf16_t* __restrict__ a1 = Xb + (rowb + 16 + lo) * DD;
    const bf16_t* __restrict__ b0 = Wm + (colb + lo) * DD;
    const bf16_t* __restrict__ b1 = Wm + (colb + 16 + lo) * DD;

    v8f c00 = {}, c01 = {}, c10 = {}, c11 = {};
#pragma unroll 2
    for (int kk = 0; kk < DD; kk += 32) {
        v16bf af0 = load_frag(a0 + kk, koff);
        v16bf af1 = load_frag(a1 + kk, koff);
        v16bf bf0 = load_frag(b0 + kk, koff);
        v16bf bf1 = load_frag(b1 + kk, koff);
        c00 = wmma_bf16(af0, bf0, c00);
        c01 = wmma_bf16(af0, bf1, c01);
        c10 = wmma_bf16(af1, bf0, c10);
        c11 = wmma_bf16(af1, bf1, c11);
    }

    const float* __restrict__ bias = (mat == 0) ? bq : (mat == 1) ? bk : bv;
    const float bc0 = bias[colb + lo];
    const float bc1 = bias[colb + 16 + lo];
    store_tile(c00, mat, rowb,      colb + lo,      bc0, koff, Qh, Kh, Vt);
    store_tile(c01, mat, rowb,      colb + 16 + lo, bc1, koff, Qh, Kh, Vt);
    store_tile(c10, mat, rowb + 16, colb + lo,      bc0, koff, Qh, Kh, Vt);
    store_tile(c11, mat, rowb + 16, colb + 16 + lo, bc1, koff, Qh, Kh, Vt);
}

// ---------------------------------------------------------------------------
// Kernel 2: flash attention. Block = 8 waves sharing one (b,h); 64-key K/V
// chunks double-buffered in LDS (rows padded to 144B = 36 dwords so 16-lane
// row-strided ds_load_b128 covers all 64 banks conflict-free), filled with
// async global->LDS copies (ASYNCcnt). Scores computed transposed
// (S^T = K*Q^T) so the softmaxed C-layout IS the A-fragment layout for P*V.
__global__ __launch_bounds__(256) void attn_kernel(
        const bf16_t* __restrict__ Qh, const bf16_t* __restrict__ Kh,
        const bf16_t* __restrict__ Vt, const float* __restrict__ mask,
        float* __restrict__ out) {
    __shared__ __align__(16) bf16_t sK[2][CHUNK][LPAD];   // [buf][key][dh]
    __shared__ __align__(16) bf16_t sV[2][DHH][LPAD];     // [buf][dh][key]

    const int tid  = threadIdx.x;
    const int wv   = tid >> 5;
    const int lane = tid & 31;
    const int lo = lane & 15, hi = lane >> 4, koff = hi << 3;

    const int bh   = blockIdx.x >> 4;           // 16 blocks per (b,h)
    const int qblk = blockIdx.x & 15;
    const int b    = bh / HH;
    const int h    = bh - b * HH;
    const int qbase = qblk * 128 + wv * 16;     // this wave's 16 q rows

    const bf16_t* __restrict__ Kbase = Kh + bh * SS * DHH;
    const bf16_t* __restrict__ Vbase = Vt + bh * DHH * SS;
    const float*  __restrict__ mrow  = mask + b * SS;
    const float LOG2E = 1.4426950408889634f;

    // Q^T B-fragments (lane = q row), loaded once.
    const bf16_t* __restrict__ qrow = Qh + (bh * SS + qbase + lo) * DHH;
    const v16bf qb0 = load_frag(qrow, koff);
    const v16bf qb1 = load_frag(qrow + 32, koff);

    v8f acc0 = {}, acc1 = {}, acc2 = {}, acc3 = {};
    float m = -1e30f, l = 0.0f;                 // per q = lane&15

    // cooperative async fill of one chunk (4 x b128 per thread)
    auto issue_chunk = [&](int buf, int kb) {
#pragma unroll
        for (int s = tid; s < 512; s += 256) {  // K: 64 rows x 128B payload
            const int row = s >> 3, part = (s & 7) << 3;
            async_copy_b128((unsigned)(uintptr_t)&sK[buf][row][part],
                            Kbase + (kb + row) * DHH + part);
        }
#pragma unroll
        for (int s = tid; s < 512; s += 256) {  // V^T: 64 dh-rows x 128B payload
            const int row = s >> 3, part = (s & 7) << 3;
            async_copy_b128((unsigned)(uintptr_t)&sV[buf][row][part],
                            Vbase + row * SS + kb + part);
        }
    };

    issue_chunk(0, 0);
    const int nchunks = SS / CHUNK;             // 32
    for (int c = 0; c < nchunks; ++c) {
        const int kb = c * CHUNK;
        const int buf = c & 1;
        if (c + 1 < nchunks) { issue_chunk(buf ^ 1, kb + CHUNK); wait_async4(); }
        else                 { wait_async0(); }
        __syncthreads();                        // chunk c resident in LDS

        const bf16_t (*K_)[LPAD] = sK[buf];
        const bf16_t (*V_)[LPAD] = sV[buf];

        // ---- S^T tiles: A = K rows (lane = key), B = Q^T ----
        v8f zero = {};
        v8f c0 = wmma_bf16(load_frag(&K_[lo     ][0],  koff), qb0, zero);
        c0     = wmma_bf16(load_frag(&K_[lo     ][32], koff), qb1, c0);
        v8f c1 = wmma_bf16(load_frag(&K_[lo + 16][0],  koff), qb0, zero);
        c1     = wmma_bf16(load_frag(&K_[lo + 16][32], koff), qb1, c1);
        v8f c2 = wmma_bf16(load_frag(&K_[lo + 32][0],  koff), qb0, zero);
        c2     = wmma_bf16(load_frag(&K_[lo + 32][32], koff), qb1, c2);
        v8f c3 = wmma_bf16(load_frag(&K_[lo + 48][0],  koff), qb0, zero);
        c3     = wmma_bf16(load_frag(&K_[lo + 48][32], koff), qb1, c3);

        // additive mask, scaled to log2 domain. lane j holds mask[kb(+32)+j]
        const float mv0 = mrow[kb + lane] * LOG2E;
        const float mv1 = mrow[kb + 32 + lane] * LOG2E;

        float s0[8], s1[8], s2[8], s3[8];
        float cm = -1e30f;
#pragma unroll
        for (int rr = 0; rr < 8; ++rr) {
            s0[rr] = c0[rr] + __shfl(mv0, rr + koff, 32);
            s1[rr] = c1[rr] + __shfl(mv0, rr + koff + 16, 32);
            s2[rr] = c2[rr] + __shfl(mv1, rr + koff, 32);
            s3[rr] = c3[rr] + __shfl(mv1, rr + koff + 16, 32);
            cm = fmaxf(cm, fmaxf(fmaxf(s0[rr], s1[rr]), fmaxf(s2[rr], s3[rr])));
        }
        cm = fmaxf(cm, __shfl_xor(cm, 16, 32));

        const float mnew = fmaxf(m, cm);
        const float sf = __builtin_exp2f(m - mnew);
        float ls = 0.0f;
        v16bf pa0, pa1;                         // P A-fragments (keys 0..31 / 32..63)
#pragma unroll
        for (int rr = 0; rr < 8; ++rr) {
            float p0 = __builtin_exp2f(s0[rr] - mnew);
            float p1 = __builtin_exp2f(s1[rr] - mnew);
            float p2 = __builtin_exp2f(s2[rr] - mnew);
            float p3 = __builtin_exp2f(s3[rr] - mnew);
            ls += (p0 + p1) + (p2 + p3);
            pa0[rr] = (bf16_t)p0; pa0[rr + 8] = (bf16_t)p1;
            pa1[rr] = (bf16_t)p2; pa1[rr + 8] = (bf16_t)p3;
        }
        ls += __shfl_xor(ls, 16, 32);
        l = l * sf + ls;
        m = mnew;

#pragma unroll
        for (int rr = 0; rr < 8; ++rr) {        // rescale: acc row q = rr+8*hi
            const float f = __shfl(sf, rr + koff, 32);
            acc0[rr] *= f; acc1[rr] *= f; acc2[rr] *= f; acc3[rr] *= f;
        }

        // ---- P*V: A = pa (lane = q), B = V^T rows (lane = dh col) ----
        acc0 = wmma_bf16(pa0, load_frag(&V_[lo     ][0],  koff), acc0);
        acc0 = wmma_bf16(pa1, load_frag(&V_[lo     ][32], koff), acc0);
        acc1 = wmma_bf16(pa0, load_frag(&V_[lo + 16][0],  koff), acc1);
        acc1 = wmma_bf16(pa1, load_frag(&V_[lo + 16][32], koff), acc1);
        acc2 = wmma_bf16(pa0, load_frag(&V_[lo + 32][0],  koff), acc2);
        acc2 = wmma_bf16(pa1, load_frag(&V_[lo + 32][32], koff), acc2);
        acc3 = wmma_bf16(pa0, load_frag(&V_[lo + 48][0],  koff), acc3);
        acc3 = wmma_bf16(pa1, load_frag(&V_[lo + 48][32], koff), acc3);

        __syncthreads();                        // protect LDS before reuse
    }

    // normalize and store: ctx element (q = rr+8*hi, dh = t*16+lo)
    const float il = 1.0f / l;
    float* __restrict__ orow = out + (b * SS + qbase) * DD + h * DHH;
#pragma unroll
    for (int rr = 0; rr < 8; ++rr) {
        const float f = __shfl(il, rr + koff, 32);
        float* __restrict__ o = orow + (rr + koff) * DD;
        o[lo     ] = acc0[rr] * f;
        o[lo + 16] = acc1[rr] * f;
        o[lo + 32] = acc2[rr] * f;
        o[lo + 48] = acc3[rr] * f;
    }
}

// ---------------------------------------------------------------------------
extern "C" void kernel_launch(void* const* d_in, const int* in_sizes, int n_in,
                              void* d_out, int out_size, void* d_ws, size_t ws_size,
                              hipStream_t stream) {
    (void)in_sizes; (void)n_in; (void)out_size; (void)ws_size;
    const float* X    = (const float*)d_in[0];
    const float* mask = (const float*)d_in[1];
    const float* Wq   = (const float*)d_in[2];
    const float* bq   = (const float*)d_in[3];
    const float* Wk   = (const float*)d_in[4];
    const float* bk   = (const float*)d_in[5];
    const float* Wv   = (const float*)d_in[6];
    const float* bv   = (const float*)d_in[7];
    float* out = (float*)d_out;

    char* ws = (char*)d_ws;
    bf16_t* Xb = (bf16_t*)(ws);                                   // 4096*768
    bf16_t* Wt = (bf16_t*)(ws + (size_t)BS * DD * 2);             // 3*768*768
    bf16_t* Qh = (bf16_t*)(ws + (size_t)BS * DD * 2 + (size_t)3 * DD * DD * 2);
    bf16_t* Kh = Qh + (size_t)BS * DD;
    bf16_t* Vt = Kh + (size_t)BS * DD;

    {
        int n = BS * DD;
        cvt_x_kernel<<<(n + 255) / 256, 256, 0, stream>>>(X, Xb, n);
    }
    {
        int n = 3 * DD * DD;
        cvt_w_kernel<<<(n + 255) / 256, 256, 0, stream>>>(Wq, Wk, Wv, Wt);
    }
    qkv_gemm_kernel<<<GEMM_WAVES / 8, 256, 0, stream>>>(Xb, Wt, bq, bk, bv, Qh, Kh, Vt);
    attn_kernel<<<BB * HH * 16, 256, 0, stream>>>(Qh, Kh, Vt, mask, out);
}
